// SQNSTDP_18743237279789
// MI455X (gfx1250) — matrix operand order repaired and checked by reading, not compiled, not run on hardware
//
#include <hip/hip_runtime.h>
#include <hip/hip_bf16.h>

typedef __attribute__((ext_vector_type(16))) _Float16 v16h;
typedef __attribute__((ext_vector_type(8)))  _Float16 v8h;
typedef __attribute__((ext_vector_type(8)))  float    v8f;

// ---------------------------------------------------------------------------
// WMMA fragment loaders (gfx1250 wave32, 16-bit operands, contiguous K order used by this code for A and B, not the ISA table order)
// A-matrix 16x32 f16: lane L holds row M=L%16; element e of v16h maps to
//   K = e + 16*(L/16)                  -> one contiguous run of 16 halfs.
// B-matrix 32x16 f16: lane L holds col N=L%16; element e maps to
//   K = e + 16*(L/16)                  -> one contiguous run of 16 halfs.
// ---------------------------------------------------------------------------
__device__ __forceinline__ v16h load_afrag(const _Float16* __restrict__ p, int half) {
  return *(const v16h*)(p + 16 * half);        // K = cb + 16*half + 0..15
}
__device__ __forceinline__ v16h load_bfrag(const _Float16* __restrict__ p, int half) {
  return *(const v16h*)(p + 16 * half);        // K = cb + 16*half + 0..15
}

// ---------------------------------------------------------------------------
// Small setup kernels
// ---------------------------------------------------------------------------
__global__ void k_init_cmax(float* cmax) {
  int i = threadIdx.x;
  if (i < 96) cmax[i] = 1e-6f;
}

__global__ void k_cvt_w(const float* __restrict__ s, _Float16* __restrict__ d,
                        int Cout, int K, int Kpad) {
  int idx = blockIdx.x * blockDim.x + threadIdx.x;
  if (idx >= Cout * Kpad) return;
  int co = idx / Kpad, k = idx - co * Kpad;
  d[idx] = (_Float16)(k < K ? s[(size_t)co * K + k] : 0.f);
}

// DoG depthwise 7x7 conv (pad 3) + per-(b,c)-plane positive max via atomicMax
__global__ void k_dog(const float* __restrict__ st, const float* __restrict__ dw,
                      float* __restrict__ dog, float* __restrict__ cmax) {
  int idx = blockIdx.x * blockDim.x + threadIdx.x;
  const int HW = 224 * 224;
  if (idx >= 32 * 3 * HW) return;
  int x = idx % 224, t = idx / 224;
  int y = t % 224;  t /= 224;
  int c = t % 3;    int b = t / 3;
  const float* plane = st + ((size_t)(b * 3 + c)) * HW;
  const float* w = dw + c * 49;
  float s = 0.f;
  #pragma unroll
  for (int ky = 0; ky < 7; ++ky) {
    int iy = y + ky - 3;
    if (iy < 0 || iy >= 224) continue;
    #pragma unroll
    for (int kx = 0; kx < 7; ++kx) {
      int ix = x + kx - 3;
      if (ix < 0 || ix >= 224) continue;
      s += plane[iy * 224 + ix] * w[ky * 7 + kx];
    }
  }
  dog[idx] = s;
  if (s > 0.f)  // positive floats order-preserve as uints
    atomicMax((unsigned int*)&cmax[b * 3 + c], __float_as_uint(s));
}

// act0 = (T - lat)*mask == 14.9 * max(dog,0)/cmax   (stored f16 for WMMA)
__global__ void k_encode(const float* __restrict__ dog, const float* __restrict__ cmax,
                         _Float16* __restrict__ act0) {
  int idx = blockIdx.x * blockDim.x + threadIdx.x;
  const int HW = 224 * 224;
  if (idx >= 32 * 3 * HW) return;
  int bc = idx / HW;
  float d = dog[idx];
  float v = d > 0.f ? 14.9f * d / cmax[bc] : 0.f;
  act0[idx] = (_Float16)v;
}

// ---------------------------------------------------------------------------
// Implicit-GEMM WMMA conv + fused winner-take-all.
// Block = one (b, y, 16-x) tile; COUT/16 waves, each computes a 16x16 D-tile.
// ---------------------------------------------------------------------------
template<int CIN, int COUT, int KSZ, int KPAD>
__global__ void k_conv_wta(const _Float16* __restrict__ act, const _Float16* __restrict__ wf,
                           int H, int W, float thr,
                           int* __restrict__ winner, float* __restrict__ val) {
  constexpr int K   = CIN * KSZ * KSZ;
  constexpr int PAD = KSZ / 2;
  constexpr int NW  = COUT / 16;
  __shared__ _Float16 ldsA[16 * KPAD];
  __shared__ float    ldsPot[16 * COUT];

  const int x0 = blockIdx.x * 16;
  const int y  = blockIdx.y;
  const int b  = blockIdx.z;
  const int tid  = threadIdx.x;          // blockDim.x == 32*NW
  const int lane = tid & 31;
  const int wave = tid >> 5;
  const int l16  = lane & 15;
  const int half = lane >> 4;

  // Stage im2col A-tile into LDS (zero-padded in K and at borders)
  for (int idx = tid; idx < 16 * KPAD; idx += 32 * NW) {
    int m = idx / KPAD;
    int k = idx - m * KPAD;
    _Float16 v = (_Float16)0.f;
    if (k < K) {
      int ci = k / (KSZ * KSZ);
      int r  = k - ci * KSZ * KSZ;
      int ky = r / KSZ, kx = r - ky * KSZ;
      int iy = y + ky - PAD;
      int ix = x0 + m + kx - PAD;
      if (iy >= 0 && iy < H && ix >= 0 && ix < W)
        v = act[(((size_t)b * CIN + ci) * H + iy) * W + ix];
    }
    ldsA[idx] = v;
  }
  __syncthreads();

  // K-loop of v_wmma_f32_16x16x32_f16
  const int nbase = wave * 16;
  const _Float16* arow = &ldsA[l16 * KPAD];
  const _Float16* brow = wf + (size_t)(nbase + l16) * KPAD;
  v8f acc = {};
  #pragma unroll 4
  for (int cb = 0; cb < KPAD; cb += 32) {
    v16h af = load_afrag(arow + cb, half);
    v16h bf = load_bfrag(brow + cb, half);
    acc = __builtin_amdgcn_wmma_f32_16x16x32_f16(false, af, false, bf,
                                                 (short)0, acc, false, false);
  }

  // Spill pot tile: D element (m=r+8*half, n=l16+nbase) in acc[r]
  #pragma unroll
  for (int r = 0; r < 8; ++r)
    ldsPot[(r + 8 * half) * COUT + nbase + l16] = acc[r];
  __syncthreads();

  // Winner-take-all per output position
  if (tid < 16) {
    int m = tid, x = x0 + m;
    if (x < W) {
      float mx = ldsPot[m * COUT];
      int   wi = 0;
      for (int c = 1; c < COUT; ++c) {
        float p = ldsPot[m * COUT + c];
        if (p > mx) { mx = p; wi = c; }   // first-max tie-break, like argmax
      }
      float v = (mx >= thr) ? (15.f * thr / (mx + 1e-5f)) : 0.f;
      size_t o = ((size_t)b * H + y) * W + x;
      winner[o] = wi;
      val[o]    = v;
    }
  }
}

// 2x2 max-pool of the (one-hot winner) spike maps; re-encode (15-c) for next
// layer (f16) or emit raw pooled spikes (f32) for the LayerNorm stage.
__global__ void k_pool(const int* __restrict__ win, const float* __restrict__ val,
                       int Cout, int Hin, int Win,
                       _Float16* __restrict__ actOut, float* __restrict__ rawOut) {
  int Hp = Hin >> 1, Wp = Win >> 1;
  long long idx = (long long)blockIdx.x * blockDim.x + threadIdx.x;
  long long total = 32LL * Hp * Wp;
  if (idx >= total) return;
  int xs = (int)(idx % Wp); long long t = idx / Wp;
  int ys = (int)(t % Hp);   int b = (int)(t / Hp);
  int w4[4]; float v4[4];
  #pragma unroll
  for (int i = 0; i < 4; ++i) {
    int y = 2 * ys + (i >> 1), x = 2 * xs + (i & 1);
    size_t o = ((size_t)b * Hin + y) * Win + x;
    w4[i] = win[o]; v4[i] = val[o];
  }
  for (int c = 0; c < Cout; ++c) {
    float m = 0.f;
    #pragma unroll
    for (int i = 0; i < 4; ++i)
      if (w4[i] == c && v4[i] > m) m = v4[i];
    size_t o = (((size_t)b * Cout + c) * Hp + ys) * Wp + xs;
    if (actOut) actOut[o] = (_Float16)(m > 0.f ? (15.f - m) : 0.f);
    else        rawOut[o] = m;
  }
}

// LayerNorm over D=50176 per sample + concat history -> padded f16 GEMM-A row
__global__ void k_ln_fx(const float* __restrict__ f, const float* __restrict__ g,
                        const float* __restrict__ bb, const float* __restrict__ hist,
                        _Float16* __restrict__ fx) {
  const int D = 50176, FPAD = 50272, HD = 90, NT = 256;
  int b = blockIdx.x, tid = threadIdx.x;
  __shared__ float s1[NT], s2[NT];
  const float* fb = f + (size_t)b * D;
  float sum = 0.f, ss = 0.f;
  for (int i = tid; i < D; i += NT) { float v = fb[i]; sum += v; ss += v * v; }
  s1[tid] = sum; s2[tid] = ss; __syncthreads();
  for (int s = NT / 2; s > 0; s >>= 1) {
    if (tid < s) { s1[tid] += s1[tid + s]; s2[tid] += s2[tid + s]; }
    __syncthreads();
  }
  float mu = s1[0] / D;
  float var = s2[0] / D - mu * mu;
  float rs = rsqrtf(var + 1e-5f);
  _Float16* fxb = fx + (size_t)b * FPAD;
  for (int i = tid; i < D; i += NT)
    fxb[i] = (_Float16)(((fb[i] - mu) * rs) * g[i] + bb[i]);
  for (int i = tid; i < FPAD - D; i += NT)
    fxb[D + i] = (_Float16)(i < HD ? hist[b * HD + i] : 0.f);
}

// ---------------------------------------------------------------------------
// K-split WMMA GEMM: each block owns one 16x16 output tile; NWAVES waves
// stride interleaved K-chunks (compile-time stride so the loop unrolls and
// loads pipeline past the WMMAs), partial D-tiles are reduced through LDS
// (deterministic fixed-order sum, no float atomics), then bias+ReLU+f16 store.
// ---------------------------------------------------------------------------
template<int NWAVES>
__global__ void k_fc_wmma(const _Float16* __restrict__ A, const _Float16* __restrict__ Wt,
                          const float* __restrict__ bias, _Float16* __restrict__ outH,
                          int KPAD, int ldo, int relu) {
  __shared__ float ldsD[NWAVES * 256];       // NWAVES x 16x16 partial tiles
  const int tid  = threadIdx.x;              // blockDim.x == 32*NWAVES
  const int lane = tid & 31;
  const int wave = tid >> 5;
  const int l16 = lane & 15, half = lane >> 4;
  const int mbase = blockIdx.x * 16, nbase = blockIdx.y * 16;
  const _Float16* arow = A  + (size_t)(mbase + l16) * KPAD;
  const _Float16* brow = Wt + (size_t)(nbase + l16) * KPAD;
  constexpr int STEP = 32 * NWAVES;
  v8f acc = {};
  #pragma unroll 4
  for (int cb = 32 * wave; cb < KPAD; cb += STEP) {
    __builtin_prefetch(arow + cb + 4 * STEP, 0, 1);   // global_prefetch_b8
    v16h af = load_afrag(arow + cb, half);
    v16h bf = load_bfrag(brow + cb, half);
    acc = __builtin_amdgcn_wmma_f32_16x16x32_f16(false, af, false, bf,
                                                 (short)0, acc, false, false);
  }
  // D element (m = r+8*half, n = l16) lives in acc[r]
  #pragma unroll
  for (int r = 0; r < 8; ++r)
    ldsD[wave * 256 + (r + 8 * half) * 16 + l16] = acc[r];
  __syncthreads();
  if (tid < 256) {                           // one thread per (m, n)
    int m = tid >> 4, n = tid & 15;
    float v = 0.f;
    #pragma unroll
    for (int w = 0; w < NWAVES; ++w) v += ldsD[w * 256 + m * 16 + n];
    v += bias[nbase + n];
    if (relu && v < 0.f) v = 0.f;
    outH[(size_t)(mbase + m) * ldo + nbase + n] = (_Float16)v;
  }
}

// Tiny final layer: (32 x 256) @ (9 x 256)^T + bias -> f32 output
__global__ void k_fc3(const _Float16* __restrict__ h2, const float* __restrict__ w,
                      const float* __restrict__ b, float* __restrict__ out) {
  int idx = blockIdx.x * blockDim.x + threadIdx.x;
  if (idx >= 32 * 9) return;
  int m = idx / 9, n = idx - m * 9;
  float s = b[n];
  for (int k = 0; k < 256; ++k)
    s += (float)h2[m * 256 + k] * w[n * 256 + k];
  out[idx] = s;
}

// ---------------------------------------------------------------------------
extern "C" void kernel_launch(void* const* d_in, const int* in_sizes, int n_in,
                              void* d_out, int out_size, void* d_ws, size_t ws_size,
                              hipStream_t stream) {
  (void)in_sizes; (void)n_in; (void)out_size; (void)ws_size;
  const float* state  = (const float*)d_in[0];
  const float* hist   = (const float*)d_in[1];
  const float* dog_w  = (const float*)d_in[2];
  const float* w1     = (const float*)d_in[3];
  const float* w2     = (const float*)d_in[4];
  const float* w3     = (const float*)d_in[5];
  const float* ln_g   = (const float*)d_in[6];
  const float* ln_b   = (const float*)d_in[7];
  const float* fw1    = (const float*)d_in[8];
  const float* fb1    = (const float*)d_in[9];
  const float* fw2    = (const float*)d_in[10];
  const float* fb2    = (const float*)d_in[11];
  const float* fw3    = (const float*)d_in[12];
  const float* fb3    = (const float*)d_in[13];
  float* out = (float*)d_out;

  size_t off = 0;
  auto alloc = [&](size_t bytes) -> void* {
    void* r = (char*)d_ws + off;
    off = (off + bytes + 255) & ~(size_t)255;
    return r;
  };
  const size_t HW0 = 224 * 224;
  float*    dog   = (float*)   alloc(32 * 3 * HW0 * 4);
  float*    cmax  = (float*)   alloc(96 * 4);
  _Float16* act0  = (_Float16*)alloc(32 * 3 * HW0 * 2);
  _Float16* w1h   = (_Float16*)alloc((size_t)32 * 96 * 2);
  _Float16* w2h   = (_Float16*)alloc((size_t)64 * 288 * 2);
  _Float16* w3h   = (_Float16*)alloc((size_t)64 * 576 * 2);
  _Float16* fw1h  = (_Float16*)alloc((size_t)128 * 50272 * 2);
  _Float16* fw2h  = (_Float16*)alloc((size_t)256 * 128 * 2);
  int*      win1  = (int*)     alloc(32 * HW0 * 4);
  float*    val1  = (float*)   alloc(32 * HW0 * 4);
  _Float16* act1  = (_Float16*)alloc((size_t)32 * 32 * 112 * 112 * 2);
  int*      win2  = (int*)     alloc((size_t)32 * 112 * 112 * 4);
  float*    val2  = (float*)   alloc((size_t)32 * 112 * 112 * 4);
  _Float16* act2  = (_Float16*)alloc((size_t)32 * 64 * 56 * 56 * 2);
  int*      win3  = (int*)     alloc((size_t)32 * 56 * 56 * 4);
  float*    val3  = (float*)   alloc((size_t)32 * 56 * 56 * 4);
  float*    out3  = (float*)   alloc((size_t)32 * 50176 * 4);
  _Float16* fx    = (_Float16*)alloc((size_t)32 * 50272 * 2);
  _Float16* h1    = (_Float16*)alloc((size_t)32 * 128 * 2);
  _Float16* h2    = (_Float16*)alloc((size_t)32 * 256 * 2);

  // -- weight conversion f32 -> f16 (K-padded) + cmax init --
  k_init_cmax<<<1, 96, 0, stream>>>(cmax);
  auto cvt = [&](const float* s, _Float16* d, int Co, int K, int Kp) {
    int total = Co * Kp;
    k_cvt_w<<<(total + 255) / 256, 256, 0, stream>>>(s, d, Co, K, Kp);
  };
  cvt(w1,  w1h,  32, 75,    96);
  cvt(w2,  w2h,  64, 288,   288);
  cvt(w3,  w3h,  64, 576,   576);
  cvt(fw1, fw1h, 128, 50266, 50272);
  cvt(fw2, fw2h, 256, 128,   128);

  // -- DoG conv + latency encode --
  {
    int total = 32 * 3 * (int)HW0;
    k_dog<<<(total + 255) / 256, 256, 0, stream>>>(state, dog_w, dog, cmax);
    k_encode<<<(total + 255) / 256, 256, 0, stream>>>(dog, cmax, act0);
  }

  // -- STDP layer 1: 3 -> 32ch @224x224, pool -> 112 --
  k_conv_wta<3, 32, 5, 96><<<dim3(14, 224, 32), 64, 0, stream>>>(
      act0, w1h, 224, 224, 15.f, win1, val1);
  { long long t = 32LL * 112 * 112;
    k_pool<<<(unsigned)((t + 255) / 256), 256, 0, stream>>>(win1, val1, 32, 224, 224, act1, nullptr); }

  // -- STDP layer 2: 32 -> 64ch @112x112, pool -> 56 --
  k_conv_wta<32, 64, 3, 288><<<dim3(7, 112, 32), 128, 0, stream>>>(
      act1, w2h, 112, 112, 12.f, win2, val2);
  { long long t = 32LL * 56 * 56;
    k_pool<<<(unsigned)((t + 255) / 256), 256, 0, stream>>>(win2, val2, 64, 112, 112, act2, nullptr); }

  // -- STDP layer 3: 64 -> 64ch @56x56, pool -> 28 (raw f32 spikes) --
  k_conv_wta<64, 64, 3, 576><<<dim3(4, 56, 32), 128, 0, stream>>>(
      act2, w3h, 56, 56, 12.f, win3, val3);
  { long long t = 32LL * 28 * 28;
    k_pool<<<(unsigned)((t + 255) / 256), 256, 0, stream>>>(win3, val3, 64, 56, 56, nullptr, out3); }

  // -- LayerNorm + history concat -> f16 GEMM-A --
  k_ln_fx<<<32, 256, 0, stream>>>(out3, ln_g, ln_b, hist, fx);

  // -- MLP head: fc1 K-split across 8 waves/block (128 waves total) --
  k_fc_wmma<8><<<dim3(2, 8),  256, 0, stream>>>(fx, fw1h, fb1, h1, 50272, 128, 1);
  k_fc_wmma<1><<<dim3(2, 16),  32, 0, stream>>>(h1, fw2h, fb2, h2, 128,   256, 1);
  k_fc3<<<2, 144, 0, stream>>>(h2, fw3, fb3, out);
}
